// LocalTimeAttention_29618094473991
// MI455X (gfx1250) — matrix operand order repaired and checked by reading, not compiled, not run on hardware
//
#include <hip/hip_runtime.h>
#include <hip/hip_fp16.h>
#include <math.h>

typedef __attribute__((ext_vector_type(16))) _Float16 v16h;
typedef __attribute__((ext_vector_type(8)))  _Float16 v8h;
typedef __attribute__((ext_vector_type(4)))  _Float16 v4h;
typedef __attribute__((ext_vector_type(8)))  float    v8f;
typedef __attribute__((ext_vector_type(4)))  unsigned int u32x4;
typedef __attribute__((ext_vector_type(4)))  int      i32x4;
typedef __attribute__((ext_vector_type(8)))  int      i32x8;

constexpr int kB  = 4;
constexpr int kT  = 2048;
constexpr int kD  = 512;
constexpr int kH  = 8;
constexpr int kHd = 64;
constexpr int kQKV = 3 * kD;   // 1536, row stride of qkv buffer

static __device__ __forceinline__ v8f vzero8() {
  v8f z;
#pragma unroll
  for (int i = 0; i < 8; ++i) z[i] = 0.0f;
  return z;
}

// ---------------------------------------------------------------------------
// TDM: DMA a 2-D f16 tile (rows x cols, row stride in elements) from global
// memory into LDS. Descriptor per CDNA5 ISA (CDNA5 ISA):
//   group0: [1:0]=count=1, [63:32]=lds_addr, [120:64]=global_addr,
//           [127:126]=type=2
//   group1: [17:16]=data_size=1 (2B), [63:48]=tensor_dim0, [95:80]=tensor_dim1,
//           [127:112]=tile_dim0, [143:128]=tile_dim1, [207:160]=dim0_stride
//   groups 2/3: zero (2-D tile). Tracked with TENSORcnt.
// ---------------------------------------------------------------------------
#if __has_builtin(__builtin_amdgcn_tensor_load_to_lds)
#define USE_TDM 1
#else
#define USE_TDM 0
#endif

#if USE_TDM
static __device__ __forceinline__ void tdm_load_2d_f16(
    const _Float16* gsrc, unsigned lds_byte_off, int cols, int rows,
    int stride_elems) {
  unsigned long long ga = (unsigned long long)(uintptr_t)gsrc;
  u32x4 g0;
  g0[0] = 1u;                                               // count = 1
  g0[1] = __builtin_amdgcn_readfirstlane(lds_byte_off);     // lds_addr
  g0[2] = __builtin_amdgcn_readfirstlane((unsigned)ga);     // gaddr[31:0]
  g0[3] = __builtin_amdgcn_readfirstlane((unsigned)(ga >> 32)) |
          0x80000000u;                                      // gaddr[56:32]|type=2
  i32x8 g1;
  g1[0] = (int)(1u << 16);                    // data_size = 1 -> 2 bytes
  g1[1] = (int)(((unsigned)cols) << 16);      // tensor_dim0[15:0]
  g1[2] = (int)(((unsigned)rows) << 16);      // tensor_dim1[15:0]
  g1[3] = (int)(((unsigned)cols) << 16);      // tile_dim0
  g1[4] = (int)((unsigned)rows);              // tile_dim1 (tile_dim2 = 0)
  g1[5] = __builtin_amdgcn_readfirstlane(stride_elems);  // dim0_stride[31:0]
  g1[6] = 0;
  g1[7] = 0;
  i32x4 gz;
  gz[0] = gz[1] = gz[2] = gz[3] = 0;
#if defined(__clang_major__) && (__clang_major__ >= 23)
  i32x8 gz8;
#pragma unroll
  for (int i = 0; i < 8; ++i) gz8[i] = 0;
  __builtin_amdgcn_tensor_load_to_lds(g0, g1, gz, gz, gz8, 0);
#else
  __builtin_amdgcn_tensor_load_to_lds(g0, g1, gz, gz, 0);
#endif
}
#endif

// ---------------------------------------------------------------------------
// fp32 -> fp16 conversion (vectorized x4)
// ---------------------------------------------------------------------------
__global__ void cvt_f32_to_f16(const float* __restrict__ src,
                               _Float16* __restrict__ dst, int n4) {
  int i = blockIdx.x * blockDim.x + threadIdx.x;
  if (i < n4) {
    float4 v = ((const float4*)src)[i];
    v4h h;
    h[0] = (_Float16)v.x; h[1] = (_Float16)v.y;
    h[2] = (_Float16)v.z; h[3] = (_Float16)v.w;
    ((v4h*)dst)[i] = h;
  }
}

// ---------------------------------------------------------------------------
// C[M,N] = A[M,K] @ W[N,K]^T + bias[N]
// 128 threads = 4 waves; block tile 64(M) x 64(N); wave tile 16(M) x 64(N).
// W tile (64 N-rows x 32 K) staged into LDS by TDM, double buffered: each
// wave DMAs its own 16-row quarter and waits on its own TENSORcnt.
// A fragment (16x32 f16): lane&15 = row M, K = 16*(lane>>4) + e
// B fragment (32x16 f16): lane&15 = col N, K = 16*(lane>>4) + e
// ---------------------------------------------------------------------------
template <bool OUT_F16>
__global__ __launch_bounds__(128) void gemm_bias_wmma(
    const _Float16* __restrict__ A, const _Float16* __restrict__ W,
    const float* __restrict__ bias, void* __restrict__ Cout,
    int M, int N, int K) {
  __shared__ _Float16 wt[2][64 * 32];      // double-buffered W tile, 8 KB

  const int lane   = threadIdx.x & 31;
  const int wave   = threadIdx.x >> 5;
  const int laneLo = lane & 15;
  const int laneHi = lane >> 4;
  const int rowBase = blockIdx.x * 64 + wave * 16;
  const int colBase = blockIdx.y * 64;

  v8f acc[4];
#pragma unroll
  for (int t = 0; t < 4; ++t) acc[t] = vzero8();

  const _Float16* arow = A + (size_t)(rowBase + laneLo) * K;

  // stage 16 rows x 32 cols of W into wt[buf] (this wave's quarter)
  auto stage = [&](int k0s, int buf) {
    const _Float16* src = W + (size_t)(colBase + wave * 16) * K + k0s;
    _Float16* dst = &wt[buf][wave * 16 * 32];
#if USE_TDM
    tdm_load_2d_f16(src, (unsigned)(uintptr_t)dst, /*cols=*/32, /*rows=*/16,
                    /*stride=*/K);
#else
    const int r = lane >> 1, half = lane & 1;   // lane -> (row, 16-half)
    *(v16h*)(dst + r * 32 + half * 16) =
        *(const v16h*)(src + (size_t)r * K + half * 16);
#endif
  };
  auto stage_wait = [&]() {
#if USE_TDM
    __builtin_amdgcn_s_wait_tensorcnt(0);
#endif
  };

  stage(0, 0);
  stage_wait();
  __syncthreads();

  for (int k0 = 0; k0 < K; k0 += 32) {
    const int buf = (k0 >> 5) & 1;
    const bool more = (k0 + 32) < K;
    if (more) stage(k0 + 32, buf ^ 1);     // prefetch next tile via TDM

    v16h a;
    {
      const _Float16* p = arow + k0 + 16 * laneHi;
      __builtin_prefetch(p + 64, 0, 1);
      v8h lo = *(const v8h*)(p);
      v8h hi = *(const v8h*)(p + 8);
#pragma unroll
      for (int i = 0; i < 8; ++i) { a[i] = lo[i]; a[i + 8] = hi[i]; }
    }
#pragma unroll
    for (int t = 0; t < 4; ++t) {
      v16h bfrag = *(const v16h*)&wt[buf][(t * 16 + laneLo) * 32 + 16 * laneHi];
      acc[t] = __builtin_amdgcn_wmma_f32_16x16x32_f16(
          false, a, false, bfrag, (short)0, acc[t], false, false);
    }

    if (more) stage_wait();                // own quarter-tile DMA done
    __syncthreads();                       // everyone's quarter visible
  }

#pragma unroll
  for (int t = 0; t < 4; ++t) {
    int col = colBase + t * 16 + laneLo;
    float bv = bias[col];
#pragma unroll
    for (int r = 0; r < 8; ++r) {
      int row = rowBase + r + 8 * laneHi;
      float v = acc[t][r] + bv;
      if (OUT_F16) ((_Float16*)Cout)[(size_t)row * N + col] = (_Float16)v;
      else         ((float*)Cout)[(size_t)row * N + col]    = v;
    }
  }
}

// ---------------------------------------------------------------------------
// Banded attention. One wave handles one (b, h, 16-query tile).
// qkv layout: [B*T, 1536] f16, cols [0,512)=Q, [512,1024)=K, [1024,1536)=V,
// each arranged as h*64 + d. Band of 48 keys starting at q0-16.
// ---------------------------------------------------------------------------
__global__ __launch_bounds__(128) void band_attn_wmma(
    const _Float16* __restrict__ qkv, const float* __restrict__ radius,
    _Float16* __restrict__ aout) {
  __shared__ _Float16 slds[4][16 * 64];

  const int lane   = threadIdx.x & 31;
  const int wave   = threadIdx.x >> 5;
  const int laneLo = lane & 15;
  const int laneHi = lane >> 4;

  const int tile = blockIdx.x * 4 + wave;         // 0 .. 4095
  const int qt = tile & 127;                      // T/16 = 128 tiles
  const int h  = (tile >> 7) & 7;
  const int b  = tile >> 10;

  const int q0 = qt * 16;
  const int k0 = q0 - 16;

  float rh = 16.0f / (1.0f + __expf(-radius[h]));
  rh = fmaxf(rh, 1.0f);

  const _Float16* base = qkv + (size_t)b * kT * kQKV;

  // --- Q fragments (A layout), K dim = head dim, 2 steps of 32 ---
  v16h qa[2];
#pragma unroll
  for (int s = 0; s < 2; ++s) {
    const _Float16* p =
        base + (size_t)(q0 + laneLo) * kQKV + h * kHd + s * 32 + 16 * laneHi;
    v8h lo = *(const v8h*)p;
    v8h hi = *(const v8h*)(p + 8);
#pragma unroll
    for (int i = 0; i < 8; ++i) { qa[s][i] = lo[i]; qa[s][i + 8] = hi[i]; }
  }

  // --- scores: Q[16x64] x K^T[64x48] -> 3 tiles of 16 cols ---
  v8f sc[3];
#pragma unroll
  for (int t = 0; t < 3; ++t) {
    sc[t] = vzero8();
    int key = k0 + t * 16 + laneLo;
    int kc  = key < 0 ? 0 : (key >= kT ? kT - 1 : key);
#pragma unroll
    for (int s = 0; s < 2; ++s) {
      const _Float16* p =
          base + (size_t)kc * kQKV + kD + h * kHd + s * 32 + 16 * laneHi;
      v16h kb = *(const v16h*)p;
      sc[t] = __builtin_amdgcn_wmma_f32_16x16x32_f16(
          false, qa[s], false, kb, (short)0, sc[t], false, false);
    }
  }

  // --- masked softmax over 48 keys. Row M = r + 8*laneHi, col = t*16+laneLo.
  //     Row reduction = shfl_xor within the 16-lane half (wave32). ---
  const float scale = 0.125f;      // 1/sqrt(64)
  const float NEGI  = -1e30f;
#pragma unroll
  for (int r = 0; r < 8; ++r) {
    int q = q0 + r + 8 * laneHi;
    float m = NEGI;
#pragma unroll
    for (int t = 0; t < 3; ++t) {
      int key = k0 + t * 16 + laneLo;
      float v = sc[t][r] * scale;
      bool ok = (key >= 0) && (key < kT) && (fabsf((float)(q - key)) <= rh);
      v = ok ? v : NEGI;
      sc[t][r] = v;
      m = fmaxf(m, v);
    }
#pragma unroll
    for (int off = 1; off < 16; off <<= 1) m = fmaxf(m, __shfl_xor(m, off, 32));
    float ssum = 0.0f;
#pragma unroll
    for (int t = 0; t < 3; ++t) {
      float p = __expf(sc[t][r] - m);
      sc[t][r] = p;
      ssum += p;
    }
#pragma unroll
    for (int off = 1; off < 16; off <<= 1) ssum += __shfl_xor(ssum, off, 32);
    float inv = 1.0f / ssum;
#pragma unroll
    for (int t = 0; t < 3; ++t) sc[t][r] *= inv;
  }

  // --- stage probs (16x48, zero-padded to 16x64) into LDS to reshape into
  //     A fragments for attn @ V ---
  _Float16* myl = slds[wave];
#pragma unroll
  for (int i = 0; i < 8; ++i) {            // zero columns 48..63
    int idx = i * 32 + lane;               // 0..255
    int rr = idx >> 4;
    int cc = idx & 15;
    myl[rr * 64 + 48 + cc] = (_Float16)0.0f;
  }
#pragma unroll
  for (int r = 0; r < 8; ++r) {
    int rr = r + 8 * laneHi;
#pragma unroll
    for (int t = 0; t < 3; ++t)
      myl[rr * 64 + t * 16 + laneLo] = (_Float16)sc[t][r];
  }
  __syncthreads();

  v16h pa[2];
#pragma unroll
  for (int s = 0; s < 2; ++s) {
    const _Float16* p = myl + laneLo * 64 + s * 32 + 16 * laneHi;
    v8h lo = *(const v8h*)p;
    v8h hi = *(const v8h*)(p + 8);
#pragma unroll
    for (int i = 0; i < 8; ++i) { pa[s][i] = lo[i]; pa[s][i + 8] = hi[i]; }
  }

  // --- O[16x64] = P[16x64] @ V[64x64] (V rows past band have zero weight) ---
#pragma unroll
  for (int nt = 0; nt < 4; ++nt) {
    v8f o = vzero8();
#pragma unroll
    for (int s = 0; s < 2; ++s) {
      v16h vb;
#pragma unroll
      for (int e = 0; e < 16; ++e) {
        int key = k0 + s * 32 + 16 * laneHi + e;
        int kc  = key < 0 ? 0 : (key >= kT ? kT - 1 : key);
        vb[e] = base[(size_t)kc * kQKV + 2 * kD + h * kHd + nt * 16 + laneLo];
      }
      o = __builtin_amdgcn_wmma_f32_16x16x32_f16(
          false, pa[s], false, vb, (short)0, o, false, false);
    }
#pragma unroll
    for (int r = 0; r < 8; ++r) {
      int row = q0 + r + 8 * laneHi;
      aout[((size_t)b * kT + row) * kD + h * kHd + nt * 16 + laneLo] =
          (_Float16)o[r];
    }
  }
}

// ---------------------------------------------------------------------------
extern "C" void kernel_launch(void* const* d_in, const int* in_sizes, int n_in,
                              void* d_out, int out_size, void* d_ws,
                              size_t ws_size, hipStream_t stream) {
  (void)in_sizes; (void)n_in; (void)out_size; (void)ws_size;

  const float* x      = (const float*)d_in[0];
  const float* radius = (const float*)d_in[1];
  const float* w_in   = (const float*)d_in[2];
  const float* b_in   = (const float*)d_in[3];
  const float* w_out  = (const float*)d_in[4];
  const float* b_out  = (const float*)d_in[5];

  char* ws = (char*)d_ws;
  size_t off = 0;
  auto alloc = [&](size_t bytes) -> void* {
    void* p = ws + off;
    off += (bytes + 255) & ~(size_t)255;
    return p;
  };
  _Float16* x_h    = (_Float16*)alloc((size_t)kB * kT * kD * 2);       // 8 MB
  _Float16* win_h  = (_Float16*)alloc((size_t)3 * kD * kD * 2);        // 1.5 MB
  _Float16* wout_h = (_Float16*)alloc((size_t)kD * kD * 2);            // 0.5 MB
  _Float16* qkv_h  = (_Float16*)alloc((size_t)kB * kT * 3 * kD * 2);   // 24 MB
  _Float16* ao_h   = (_Float16*)alloc((size_t)kB * kT * kD * 2);       // 8 MB

  int n4;
  n4 = kB * kT * kD / 4;
  cvt_f32_to_f16<<<(n4 + 255) / 256, 256, 0, stream>>>(x, x_h, n4);
  n4 = 3 * kD * kD / 4;
  cvt_f32_to_f16<<<(n4 + 255) / 256, 256, 0, stream>>>(w_in, win_h, n4);
  n4 = kD * kD / 4;
  cvt_f32_to_f16<<<(n4 + 255) / 256, 256, 0, stream>>>(w_out, wout_h, n4);

  // QKV: [8192,512] x [512,1536] + bias -> f16
  gemm_bias_wmma<true><<<dim3(kB * kT / 64, 3 * kD / 64), 128, 0, stream>>>(
      x_h, win_h, b_in, qkv_h, kB * kT, 3 * kD, kD);

  // banded attention: 4096 (b,h,qtile) waves, 4 per block
  band_attn_wmma<<<kB * kH * (kT / 16) / 4, 128, 0, stream>>>(qkv_h, radius,
                                                              ao_h);

  // out projection: [8192,512] x [512,512] + bias -> f32 d_out
  gemm_bias_wmma<false><<<dim3(kB * kT / 64, kD / 64), 128, 0, stream>>>(
      ao_h, wout_h, b_out, (float*)d_out, kB * kT, kD, kD);
}
